// GNN_GIN_74285754351849
// MI455X (gfx1250) — matrix operand order repaired and checked by reading, not compiled, not run on hardware
//
#include <hip/hip_runtime.h>
#include <hip/hip_bf16.h>

#define NN 100000   // nodes
#define NE 150000   // edges
#define D  512      // feature dim (D_IN == D_HID)
#define NG 64       // graphs

typedef __attribute__((ext_vector_type(16))) __bf16    v16bf;
typedef __attribute__((ext_vector_type(8)))  float     v8f;
typedef __attribute__((ext_vector_type(4)))  unsigned  uint4v;

union Frag { struct { uint4v lo, hi; } u; v16bf v; };

// pack two f32 -> packed bf16 pair (RNE), low word = first element
__device__ __forceinline__ unsigned pack_bf16(float a, float b) {
#if __has_builtin(__builtin_amdgcn_cvt_pk_bf16_f32)
    auto r = __builtin_amdgcn_cvt_pk_bf16_f32(a, b);
    return __builtin_bit_cast(unsigned, r);
#else
    union { float f; unsigned u; } x, y; x.f = a; y.f = b;
    unsigned ra = (x.u + 0x7FFFu + ((x.u >> 16) & 1u)) >> 16;
    unsigned rb = (y.u + 0x7FFFu + ((y.u >> 16) & 1u)) >> 16;
    return (ra & 0xFFFFu) | (rb << 16);
#endif
}

// ---- vectorized f32 copy (agg init: z = agg + x with eps=0) ---------------
__global__ void copy_f32(const float* __restrict__ src, float* __restrict__ dst, long n4) {
    long i = (long)blockIdx.x * blockDim.x + threadIdx.x;
    long stride = (long)gridDim.x * blockDim.x;
    const float4* s = (const float4*)src;
    float4* d = (float4*)dst;
    for (; i < n4; i += stride) d[i] = s[i];
}

// ---- edge scatter-add: agg[dst] += x[src]  (128 lanes x float4 per edge) --
__global__ void edge_scatter(const float* __restrict__ x, const int* __restrict__ ei,
                             float* __restrict__ agg) {
    long g = (long)blockIdx.x * blockDim.x + threadIdx.x;  // g = edge*128 + t
    if (g >= (long)NE * 128) return;
    int e = (int)(g >> 7);
    int t = (int)(g & 127);
    int src = ei[e];
    int dst = ei[NE + e];
    float4 v = *(const float4*)(x + (long)src * D + t * 4);
    float* p = agg + (long)dst * D + t * 4;
    unsafeAtomicAdd(p + 0, v.x);
    unsafeAtomicAdd(p + 1, v.y);
    unsafeAtomicAdd(p + 2, v.z);
    unsafeAtomicAdd(p + 3, v.w);
}

// ---- one-time weight packing: Wp[kp][col] = pack(W[2kp][col], W[2kp+1][col])
// 512x512 f32 -> 256x512 packed bf16 pairs (512 KB, L2-resident).
__global__ void pack_weights(const float* __restrict__ W, unsigned* __restrict__ Wp) {
    int idx = blockIdx.x * 256 + threadIdx.x;       // 256*512 items
    if (idx >= (D / 2) * D) return;
    int kp = idx >> 9, col = idx & (D - 1);
    Wp[idx] = pack_bf16(W[(long)(2 * kp) * D + col], W[(long)(2 * kp + 1) * D + col]);
}

// ---- tiled WMMA GEMM: C[M x 512] = relu(A[M x 512] @ W[512 x 512] + bias) -
// Block tile 128x64, 256 threads = 8 waves; each wave owns a 32x32 sub-tile
// (2x2 v_wmma_f32_16x16x32_bf16 accumulators). BK = 32, double-buffered LDS,
// software-pipelined: global loads for tile k+1 overlap WMMAs of tile k.
// Grid is (n-blocks, m-blocks) with n FASTEST so the 8 blocks sharing the same
// A rows launch adjacently -> A re-reads hit L2, HBM sees A ~once.
// LDS: bf16 pairs packed along K; pitch 20 dwords keeps 16B alignment.
//   As_u[buf][row][kp] : A row-major (128 x 16 pairs), packed on the fly
//   Bt_u[buf][col][kp] : B col-major ( 64 x 16 pairs), pre-packed from Wp
#define BM 128
#define BN 64
#define BK 32
#define NK (D / BK)   // 16 K-steps
#define PITCH 20

__global__ __launch_bounds__(256) void gemm_bias_relu(
    const float* __restrict__ A, const unsigned* __restrict__ Wp,
    const float* __restrict__ bias, float* __restrict__ C, int M)
{
    __shared__ unsigned As_u[2][BM][PITCH];
    __shared__ unsigned Bt_u[2][BN][PITCH];

    const int tid  = threadIdx.x;
    const int lane = tid & 31;
    const int wid  = tid >> 5;
    const int wm   = (wid & 3) * 32;   // wave row offset inside block tile
    const int wn   = (wid >> 2) * 32;  // wave col offset inside block tile
    const int n0   = blockIdx.x * BN;  // n fastest-varying (L2 reuse of A)
    const int m0   = blockIdx.y * BM;

    // ---- per-thread staging geometry (loop-invariant) ----
    // A: 2048 pairs/tile, 8 per thread; item i -> row (tid>>4)+i*16, pair tid&15
    const int aR0 = tid >> 4, aCp = tid & 15;
    const float2* aSrc[8];
    unsigned aMask = 0;
    #pragma unroll
    for (int i = 0; i < 8; ++i) {
        int row  = m0 + aR0 + i * 16;
        bool ok  = row < M;
        if (ok) aMask |= (1u << i);
        int rowc = ok ? row : (M - 1);               // clamp: branch-free loads
        aSrc[i]  = (const float2*)(A + (long)rowc * D) + aCp;
    }
    // B: 1024 pre-packed pairs/tile, 4 per thread; item i -> col tid&63, rp (tid>>6)+i*4
    const int bC = tid & 63, bRp0 = tid >> 6;
    const unsigned* bSrc[4];
    #pragma unroll
    for (int i = 0; i < 4; ++i)
        bSrc[i] = Wp + (long)(bRp0 + i * 4) * D + n0 + bC;   // advance (BK/2)*D per step

    // ---- fragment-native K offsets (pair units) ----
    const int kpA = (lane < 16) ? 0 : 4;   // A: lanes 16-31 start at K=8
    const int kpB = (lane < 16) ? 0 : 4;   // B: lanes 16-31 start at K=8
    const int l15 = lane & 15;

    v8f acc[2][2] = {};

    // ---- prologue: stage tile 0 into buffer 0 ----
    #pragma unroll
    for (int i = 0; i < 8; ++i) {
        float2 v = aSrc[i][0];
        float a = (aMask >> i) & 1 ? v.x : 0.0f;
        float b = (aMask >> i) & 1 ? v.y : 0.0f;
        As_u[0][aR0 + i * 16][aCp] = pack_bf16(a, b);
    }
    #pragma unroll
    for (int i = 0; i < 4; ++i)
        Bt_u[0][bC][bRp0 + i * 4] = bSrc[i][0];
    __syncthreads();

    for (int kt = 0; kt < NK; ++kt) {
        const int cur = kt & 1;
        const int nxt = cur ^ 1;

        // 1) issue next tile's global loads early (overlap with WMMA below)
        float2   aReg[8];
        unsigned bReg[4];
        if (kt + 1 < NK) {
            const int ko = (kt + 1) * (BK / 2);            // float2 offset for A
            #pragma unroll
            for (int i = 0; i < 8; ++i) aReg[i] = aSrc[i][ko];
            const long wo = (long)(kt + 1) * (BK / 2) * D; // uint offset for Wp
            #pragma unroll
            for (int i = 0; i < 4; ++i) bReg[i] = bSrc[i][wo];
        }

        // 2) fragments from current buffer + 4 WMMAs
        Frag af[2], bf_[2];
        #pragma unroll
        for (int am = 0; am < 2; ++am) {
            int row = wm + am * 16 + l15;
            af[am].u.lo = *(const uint4v*)&As_u[cur][row][kpA];
            af[am].u.hi = *(const uint4v*)&As_u[cur][row][kpA + 8];
        }
        #pragma unroll
        for (int bn = 0; bn < 2; ++bn) {
            int col = wn + bn * 16 + l15;
            bf_[bn].u.lo = *(const uint4v*)&Bt_u[cur][col][kpB];
            bf_[bn].u.hi = *(const uint4v*)&Bt_u[cur][col][kpB + 8];
        }
        #pragma unroll
        for (int am = 0; am < 2; ++am)
            #pragma unroll
            for (int bn = 0; bn < 2; ++bn)
                acc[am][bn] = __builtin_amdgcn_wmma_f32_16x16x32_bf16(
                    false, af[am].v, false, bf_[bn].v,
                    (short)0, acc[am][bn], false, false);

        // 3) pack + store next tile into the other buffer
        if (kt + 1 < NK) {
            #pragma unroll
            for (int i = 0; i < 8; ++i) {
                float a = (aMask >> i) & 1 ? aReg[i].x : 0.0f;
                float b = (aMask >> i) & 1 ? aReg[i].y : 0.0f;
                As_u[nxt][aR0 + i * 16][aCp] = pack_bf16(a, b);
            }
            #pragma unroll
            for (int i = 0; i < 4; ++i)
                Bt_u[nxt][bC][bRp0 + i * 4] = bReg[i];
        }
        __syncthreads();
    }

    // ---- epilogue: C/D layout lane<16: M=r, lane>=16: M=8+r; N=lane%16 ----
    const int rBase = wm + ((lane >> 4) << 3);
    if (m0 + BM <= M) {
        // fast path: full tile, no guards (781 of 782 m-blocks)
        #pragma unroll
        for (int am = 0; am < 2; ++am) {
            #pragma unroll
            for (int bn = 0; bn < 2; ++bn) {
                int col = n0 + wn + bn * 16 + l15;
                float b = bias[col];
                float* cp = C + (long)(m0 + rBase + am * 16) * D + col;
                #pragma unroll
                for (int r = 0; r < 8; ++r)
                    cp[(long)r * D] = fmaxf(acc[am][bn][r] + b, 0.0f);
            }
        }
    } else {
        #pragma unroll
        for (int am = 0; am < 2; ++am) {
            #pragma unroll
            for (int bn = 0; bn < 2; ++bn) {
                int col = n0 + wn + bn * 16 + l15;
                float b = bias[col];
                #pragma unroll
                for (int r = 0; r < 8; ++r) {
                    int row = m0 + rBase + am * 16 + r;
                    if (row < M)
                        C[(long)row * D + col] = fmaxf(acc[am][bn][r] + b, 0.0f);
                }
            }
        }
    }
}

// ---- per-node head: relu(dot(agg2[n], W2a)+b2a)*W2b+b2b -> graph pools ----
__global__ __launch_bounds__(256) void node_head(
    const float* __restrict__ agg, const float* __restrict__ w2a,
    const float* __restrict__ b2a, const float* __restrict__ w2b,
    const float* __restrict__ b2b, const int* __restrict__ batch,
    float* __restrict__ gsum, float* __restrict__ gcnt)
{
    int wid  = threadIdx.x >> 5;
    int lane = threadIdx.x & 31;
    int node = blockIdx.x * 8 + wid;
    if (node >= NN) return;
    const float4* row = (const float4*)(agg + (long)node * D);
    const float4* w   = (const float4*)w2a;
    float s = 0.0f;
    #pragma unroll
    for (int i = 0; i < 4; ++i) {   // 32 lanes x 16 floats = 512
        float4 a = row[lane * 4 + i];
        float4 b = w[lane * 4 + i];
        s += a.x * b.x + a.y * b.y + a.z * b.z + a.w * b.w;
    }
    #pragma unroll
    for (int off = 16; off > 0; off >>= 1) s += __shfl_down(s, off, 32);
    if (lane == 0) {
        float z   = fmaxf(s + b2a[0], 0.0f);
        float val = z * w2b[0] + b2b[0];
        int g = batch[node];
        unsafeAtomicAdd(&gsum[g], val);
        unsafeAtomicAdd(&gcnt[g], 1.0f);
    }
}

__global__ void zero128(float* p) { if (threadIdx.x < 128) p[threadIdx.x] = 0.0f; }

__global__ void finalize(const float* __restrict__ gsum, const float* __restrict__ gcnt,
                         float* __restrict__ out) {
    int g = threadIdx.x;
    if (g < NG) out[g] = gsum[g] / fmaxf(gcnt[g], 1.0f);
}

// ---------------------------------------------------------------------------
extern "C" void kernel_launch(void* const* d_in, const int* in_sizes, int n_in,
                              void* d_out, int out_size, void* d_ws, size_t ws_size,
                              hipStream_t stream) {
    const float* x   = (const float*)d_in[0];
    const int*   ei  = (const int*)d_in[1];
    const int*   bat = (const int*)d_in[2];
    const float* W1a = (const float*)d_in[3];
    const float* b1a = (const float*)d_in[4];
    const float* W1b = (const float*)d_in[5];
    const float* b1b = (const float*)d_in[6];
    const float* W2a = (const float*)d_in[7];
    const float* b2a = (const float*)d_in[8];
    const float* W2b = (const float*)d_in[9];
    const float* b2b = (const float*)d_in[10];
    float* out = (float*)d_out;

    const size_t featBytes = (size_t)NN * D * sizeof(float);  // 204.8 MB
    const size_t wpBytes   = (size_t)(D / 2) * D * sizeof(unsigned); // 512 KB
    char*     ws   = (char*)d_ws;
    float*    bufA = (float*)ws;
    float*    bufB = (float*)(ws + featBytes);
    unsigned* Wp1  = (unsigned*)(ws + 2 * featBytes);
    unsigned* Wp2  = (unsigned*)(ws + 2 * featBytes + wpBytes);
    float*    gsum = (float*)(ws + 2 * featBytes + 2 * wpBytes);
    float*    gcnt = gsum + NG;

    const long n4 = (long)NN * D / 4;
    dim3 gGemm(D / BN, (NN + BM - 1) / BM);    // n fastest -> A reuse in L2
    const int scatterBlocks = (int)(((long)NE * 128 + 255) / 256);

    // One-time (per launch) weight packing into bf16-pair layout.
    pack_weights<<<(D / 2) * D / 256, 256, 0, stream>>>(W1a, Wp1);
    pack_weights<<<(D / 2) * D / 256, 256, 0, stream>>>(W1b, Wp2);

    // Layer 1 aggregation: bufA = x ; bufA[dst] += x[src]
    copy_f32<<<2048, 256, 0, stream>>>(x, bufA, n4);
    edge_scatter<<<scatterBlocks, 256, 0, stream>>>(x, ei, bufA);
    // MLP1: bufB = relu(bufA @ W1a + b1a) ; bufA = relu(bufB @ W1b + b1b)
    gemm_bias_relu<<<gGemm, 256, 0, stream>>>(bufA, Wp1, b1a, bufB, NN);
    gemm_bias_relu<<<gGemm, 256, 0, stream>>>(bufB, Wp2, b1b, bufA, NN);
    // Layer 2 aggregation: bufB = h ; bufB[dst] += h[src]
    copy_f32<<<2048, 256, 0, stream>>>(bufA, bufB, n4);
    edge_scatter<<<scatterBlocks, 256, 0, stream>>>(bufA, ei, bufB);
    // Head + mean pool
    zero128<<<1, 128, 0, stream>>>(gsum);
    node_head<<<(NN + 7) / 8, 256, 0, stream>>>(bufB, W2a, b2a, W2b, b2b, bat, gsum, gcnt);
    finalize<<<1, 64, 0, stream>>>(gsum, gcnt, out);
}
